// MultiHeadAttention_46712064311738
// MI455X (gfx1250) — compile-verified, matrix operand order checked by reading, not run on hardware
//
#include <hip/hip_runtime.h>
#include <hip/hip_bf16.h>
#include <cstdint>

typedef __attribute__((ext_vector_type(16))) __bf16 v16bf;
typedef __attribute__((ext_vector_type(8)))  float  v8f;
typedef int v4i __attribute__((vector_size(16)));

constexpr int BB = 4, TT = 2048, CC = 1024, HH = 16, DD = 64;
constexpr int MM = BB * TT;   // 8192 rows
constexpr int N3 = 3 * CC;    // 3072

// ---------------------------------------------------------------------------
// Async global->LDS staging (CDNA5 GLOBAL_LOAD_ASYNC_TO_LDS_B128, ASYNCcnt)
// Tier 2: clang builtin; Tier 1: inline asm; Tier 0: sync copy fallback.
// ---------------------------------------------------------------------------
#if __has_builtin(__builtin_amdgcn_global_load_async_to_lds_b128) && \
    __has_builtin(__builtin_amdgcn_s_wait_asynccnt)
#define ASYNC_MODE 2
#else
#define ASYNC_MODE 1
#endif

__device__ __forceinline__ void async_copy16(const void* g, void* l) {
#if ASYNC_MODE == 2
  __builtin_amdgcn_global_load_async_to_lds_b128(
      (__attribute__((address_space(1))) v4i*)(uintptr_t)g,
      (__attribute__((address_space(3))) v4i*)(unsigned)(uintptr_t)l, 0, 0);
#elif ASYNC_MODE == 1
  unsigned lds_off = (unsigned)(uintptr_t)l;   // LDS byte offset (addr[31:0])
  asm volatile("global_load_async_to_lds_b128 %0, %1, off"
               :: "v"(lds_off), "v"((unsigned long long)(uintptr_t)g)
               : "memory");
#else
  *(uint4*)l = *(const uint4*)g;
#endif
}

template <int N>
__device__ __forceinline__ void wait_async() {
#if ASYNC_MODE == 2
  __builtin_amdgcn_s_wait_asynccnt(N);
#elif ASYNC_MODE == 1
  if constexpr (N == 0)
    asm volatile("s_wait_asynccnt 0x0" ::: "memory");
  else
    asm volatile("s_wait_asynccnt 0x4" ::: "memory");
#endif
}

// ---------------------------------------------------------------------------
// WMMA helpers (CDNA5 16x16x32 bf16, f32 accumulate)
// ---------------------------------------------------------------------------
__device__ __forceinline__ v8f wmma_bf16(v16bf a, v16bf b, v8f c) {
  return __builtin_amdgcn_wmma_f32_16x16x32_bf16(false, a, false, b, (short)0, c,
                                                 false, false);
}

// A-fragment (16x32, 16-bit): lane L<16 holds row M=L, K in [0,8)+[16,24);
// lane L>=16 holds row M=L-16, K in [8,16)+[24,32).  Two contiguous 16B runs.
__device__ __forceinline__ v16bf lds_frag_a(const __bf16* base, int stride,
                                            int mrow, int k0) {
  const int lane = threadIdx.x & 31;
  const __bf16* p = base + (mrow + (lane & 15)) * stride + k0 + ((lane >> 4) << 3);
  v16bf a;
#pragma unroll
  for (int i = 0; i < 8; ++i) a[i] = p[i];
#pragma unroll
  for (int i = 0; i < 8; ++i) a[i + 8] = p[16 + i];
  return a;
}

// B-fragment (32x16, 16-bit) from row-major [K][N] LDS tile: lane = column N,
// elements walk K (mirrors A layout with N in place of M).
__device__ __forceinline__ v16bf lds_frag_b(const __bf16* base, int stride,
                                            int ncol) {
  const int lane = threadIdx.x & 31;
  const int kb = (lane >> 4) << 3;
  const __bf16* p = base + ncol + (lane & 15);
  v16bf b;
#pragma unroll
  for (int i = 0; i < 8; ++i) b[i] = p[(kb + i) * stride];
#pragma unroll
  for (int i = 0; i < 8; ++i) b[i + 8] = p[(kb + 16 + i) * stride];
  return b;
}

// A-fragment straight from a global [rows][64] bf16 matrix (Q tiles).
__device__ __forceinline__ v16bf glb_frag_a64(const __bf16* base, size_t row0,
                                              int k0) {
  const int lane = threadIdx.x & 31;
  const __bf16* p = base + (row0 + (size_t)(lane & 15)) * DD + k0 + ((lane >> 4) << 3);
  v16bf a;
#pragma unroll
  for (int i = 0; i < 8; ++i) a[i] = p[i];
#pragma unroll
  for (int i = 0; i < 8; ++i) a[i + 8] = p[16 + i];
  return a;
}

// ---------------------------------------------------------------------------
// f32 -> bf16 conversion
// ---------------------------------------------------------------------------
__global__ void cvt_f32_bf16(const float* __restrict__ in,
                             __bf16* __restrict__ out, int n) {
  int i = blockIdx.x * blockDim.x + threadIdx.x;
  if (i < n) out[i] = (__bf16)in[i];
}

// ---------------------------------------------------------------------------
// Tiled bf16 WMMA GEMM: C[M,N] = A[M,K] * B[K,N] (+bias).
// 256 threads = 8 waves; block tile 128x128, BK=32; wave tile 64x32.
// Double-buffered LDS staged via async global->LDS loads.
// MODE 0: QKV epilogue (split into Q/K/V [B,H,T,D] bf16, Q scaled by 1/sqrt(D))
// MODE 1: f32 output epilogue (projection)
// ---------------------------------------------------------------------------
template <int MODE>
__global__ __launch_bounds__(256) void gemm_bf16_wmma(
    const __bf16* __restrict__ A, const __bf16* __restrict__ Bm,
    const float* __restrict__ bias, __bf16* __restrict__ Qo,
    __bf16* __restrict__ Ko, __bf16* __restrict__ Vo, float* __restrict__ Out,
    int Kdim, int Ndim) {
  __shared__ __bf16 As[2][128][48];   // 96B rows (16B aligned)
  __shared__ __bf16 Bs[2][32][144];   // 288B rows (16B aligned)

  const int tid = threadIdx.x;
  const int lane = tid & 31;
  const int wave = tid >> 5;
  const int half = lane >> 4;
  const int wm = (wave >> 2) * 64;  // wave row offset in block tile
  const int wn = (wave & 3) * 32;   // wave col offset in block tile
  const int bm = blockIdx.y * 128;
  const int bn = blockIdx.x * 128;

  // Stage one 128x32 A tile + 32x128 B tile into LDS buffer `buf`.
  // 4 x 16B async copies per thread.
  auto stage = [&](int buf, int k0) {
#pragma unroll
    for (int it = 0; it < 2; ++it) {
      int idx = tid + it * 256;                 // 512 chunks of 8 bf16
      int r = idx >> 2, c = (idx & 3) * 8;
      async_copy16(A + (size_t)(bm + r) * Kdim + k0 + c, &As[buf][r][c]);
    }
#pragma unroll
    for (int it = 0; it < 2; ++it) {
      int idx = tid + it * 256;
      int r = idx >> 4, c = (idx & 15) * 8;
      async_copy16(Bm + (size_t)(k0 + r) * Ndim + bn + c, &Bs[buf][r][c]);
    }
  };

  v8f acc[4][2] = {};
  const int nk = Kdim / 32;

  stage(0, 0);
  for (int k = 0; k < nk; ++k) {
    const int cur = k & 1;
    if (k + 1 < nk) {
      stage(cur ^ 1, (k + 1) * 32);   // prefetch next tile into other buffer
      wait_async<4>();                // tile k complete (in-order), k+1 in flight
    } else {
      wait_async<0>();
    }
    __syncthreads();

    v16bf af[4], bfr[2];
#pragma unroll
    for (int mi = 0; mi < 4; ++mi)
      af[mi] = lds_frag_a(&As[cur][0][0], 48, wm + mi * 16, 0);
#pragma unroll
    for (int ni = 0; ni < 2; ++ni)
      bfr[ni] = lds_frag_b(&Bs[cur][0][0], 144, wn + ni * 16);
#pragma unroll
    for (int mi = 0; mi < 4; ++mi)
#pragma unroll
      for (int ni = 0; ni < 2; ++ni)
        acc[mi][ni] = wmma_bf16(af[mi], bfr[ni], acc[mi][ni]);
    __syncthreads();  // all reads of buf `cur` done before it is re-staged
  }

  // Epilogue.  C/D layout: VGPR r -> row r + 8*half, lane&15 -> col.
  constexpr float qscale = 0.125f;  // 1/sqrt(64)
#pragma unroll
  for (int mi = 0; mi < 4; ++mi) {
#pragma unroll
    for (int ni = 0; ni < 2; ++ni) {
#pragma unroll
      for (int r = 0; r < 8; ++r) {
        int gm = bm + wm + mi * 16 + r + half * 8;
        int gn = bn + wn + ni * 16 + (lane & 15);
        float v = acc[mi][ni][r] + bias[gn];
        if (MODE == 0) {
          int sel = gn >> 10;       // 0=Q 1=K 2=V
          int cc = gn & (CC - 1);
          int h = cc >> 6, d = cc & 63;
          int b = gm >> 11, t = gm & (TT - 1);
          size_t o = ((((size_t)b * HH) + h) * TT + t) * DD + d;
          if (sel == 0)
            Qo[o] = (__bf16)(v * qscale);
          else if (sel == 1)
            Ko[o] = (__bf16)v;
          else
            Vo[o] = (__bf16)v;
        } else {
          Out[(size_t)gm * Ndim + gn] = v;
        }
      }
    }
  }
}

// ---------------------------------------------------------------------------
// Flash attention: grid (T/64, B*H), 128 threads = 4 waves.
// Each wave owns 16 query rows; 32-key tiles; online softmax.
// K/V tiles staged via async global->LDS loads.
// ---------------------------------------------------------------------------
__global__ __launch_bounds__(128) void attn_wmma(
    const __bf16* __restrict__ Qb, const __bf16* __restrict__ Kb,
    const __bf16* __restrict__ Vb, __bf16* __restrict__ Yb) {
  __shared__ __bf16 Kt[32][80];        // key tile [key][d], 160B rows
  __shared__ __bf16 Vt[32][80];
  __shared__ __bf16 Pls[4][16][48];    // per-wave P relayout scratch

  const int tid = threadIdx.x;
  const int lane = tid & 31;
  const int wave = tid >> 5;
  const int half = lane >> 4;
  const int bh = blockIdx.y;           // 0..63  (b*16+h)
  const int q0b = blockIdx.x * 64;     // block query base
  const int q0 = q0b + wave * 16;      // wave query base
  const size_t rowbase = (size_t)bh * TT;

  v16bf qf[2];
  qf[0] = glb_frag_a64(Qb, rowbase + q0, 0);
  qf[1] = glb_frag_a64(Qb, rowbase + q0, 32);

  v8f O[4] = {};
  float mrow[8], lrow[8];
#pragma unroll
  for (int r = 0; r < 8; ++r) { mrow[r] = -1e30f; lrow[r] = 0.0f; }

  const int nkb = (q0b + 64) / 32;     // causal: keys <= q0b+63
  for (int kb = 0; kb < nkb; ++kb) {
    const int kbase0 = kb * 32;
    // Async K/V tile load: 32x64 bf16 each, 4 x 16B copies per thread
#pragma unroll
    for (int it = 0; it < 2; ++it) {
      int idx = tid + it * 128;
      int r = idx >> 3, c = (idx & 7) * 8;
      async_copy16(Kb + (rowbase + kbase0 + r) * DD + c, &Kt[r][c]);
      async_copy16(Vb + (rowbase + kbase0 + r) * DD + c, &Vt[r][c]);
    }
    wait_async<0>();
    __syncthreads();

    // S = Q * K^T  (two 16-key column tiles, contraction over D in 2 chunks)
    v8f S[2] = {};
#pragma unroll
    for (int ni = 0; ni < 2; ++ni)
#pragma unroll
      for (int ci = 0; ci < 2; ++ci) {
        v16bf kfrag = lds_frag_a(&Kt[0][0], 80, ni * 16, ci * 32);
        S[ni] = wmma_bf16(qf[ci], kfrag, S[ni]);
      }

    // Causal mask + online softmax (row = r + 8*half, col = lane&15)
#pragma unroll
    for (int r = 0; r < 8; ++r) {
      const int q = q0 + r + half * 8;
#pragma unroll
      for (int ni = 0; ni < 2; ++ni) {
        int k = kbase0 + ni * 16 + (lane & 15);
        if (k > q) S[ni][r] = -1e30f;
      }
      float mx = fmaxf(S[0][r], S[1][r]);
#pragma unroll
      for (int off = 8; off >= 1; off >>= 1)
        mx = fmaxf(mx, __shfl_xor(mx, off, 16));
      float mnew = fmaxf(mrow[r], mx);
      float corr = __expf(mrow[r] - mnew);
      mrow[r] = mnew;
      float p0 = __expf(S[0][r] - mnew);
      float p1 = __expf(S[1][r] - mnew);
      float rs = p0 + p1;
#pragma unroll
      for (int off = 8; off >= 1; off >>= 1) rs += __shfl_xor(rs, off, 16);
      lrow[r] = lrow[r] * corr + rs;
#pragma unroll
      for (int di = 0; di < 4; ++di) O[di][r] *= corr;
      // relayout P into A-fragment via per-wave LDS (DS ops in-order per wave)
      Pls[wave][r + half * 8][(lane & 15)] = (__bf16)p0;
      Pls[wave][r + half * 8][16 + (lane & 15)] = (__bf16)p1;
    }
    __builtin_amdgcn_wave_barrier();

    // O += P * V   (N = D in four 16-col tiles, contraction over 32 keys)
    v16bf pf = lds_frag_a(&Pls[wave][0][0], 48, 0, 0);
#pragma unroll
    for (int di = 0; di < 4; ++di) {
      v16bf vfrag = lds_frag_b(&Vt[0][0], 80, di * 16);
      O[di] = wmma_bf16(pf, vfrag, O[di]);
    }
    __syncthreads();
  }

  // Normalize and write y -> [B, T, C] bf16 (col = h*64 + d)
  const int b = bh >> 4;
  const int h = bh & 15;
#pragma unroll
  for (int r = 0; r < 8; ++r) {
    const int q = q0 + r + half * 8;
    const float inv = 1.0f / lrow[r];
    const size_t row = ((size_t)b * TT + q) * CC + h * 64;
#pragma unroll
    for (int di = 0; di < 4; ++di)
      Yb[row + di * 16 + (lane & 15)] = (__bf16)(O[di][r] * inv);
  }
}

// ---------------------------------------------------------------------------
// Host launcher
// ---------------------------------------------------------------------------
extern "C" void kernel_launch(void* const* d_in, const int* in_sizes, int n_in,
                              void* d_out, int out_size, void* d_ws,
                              size_t ws_size, hipStream_t stream) {
  const float* x      = (const float*)d_in[0];
  const float* w_qkv  = (const float*)d_in[1];
  const float* b_qkv  = (const float*)d_in[2];
  const float* w_proj = (const float*)d_in[3];
  const float* b_proj = (const float*)d_in[4];
  float* out = (float*)d_out;

  size_t off = 0;
  auto alloc = [&](size_t bytes) -> void* {
    void* p = (char*)d_ws + off;
    off += (bytes + 255) & ~(size_t)255;
    return p;
  };
  __bf16* Xbf  = (__bf16*)alloc((size_t)MM * CC * 2);
  __bf16* Wqkv = (__bf16*)alloc((size_t)CC * N3 * 2);
  __bf16* Wprj = (__bf16*)alloc((size_t)CC * CC * 2);
  __bf16* Qbf  = (__bf16*)alloc((size_t)MM * CC * 2);
  __bf16* Kbf  = (__bf16*)alloc((size_t)MM * CC * 2);
  __bf16* Vbf  = (__bf16*)alloc((size_t)MM * CC * 2);
  __bf16* Ybf  = (__bf16*)alloc((size_t)MM * CC * 2);

  const int cvtB = 256;
  cvt_f32_bf16<<<(MM * CC + cvtB - 1) / cvtB, cvtB, 0, stream>>>(x, Xbf, MM * CC);
  cvt_f32_bf16<<<(CC * N3 + cvtB - 1) / cvtB, cvtB, 0, stream>>>(w_qkv, Wqkv, CC * N3);
  cvt_f32_bf16<<<(CC * CC + cvtB - 1) / cvtB, cvtB, 0, stream>>>(w_proj, Wprj, CC * CC);

  // QKV GEMM: [8192 x 3072] = Xbf[8192x1024] * Wqkv[1024x3072]
  gemm_bf16_wmma<0><<<dim3(N3 / 128, MM / 128), 256, 0, stream>>>(
      Xbf, Wqkv, b_qkv, Qbf, Kbf, Vbf, nullptr, CC, N3);

  // Flash attention: grid (T/64, B*H)
  attn_wmma<<<dim3(TT / 64, BB * HH), 128, 0, stream>>>(Qbf, Kbf, Vbf, Ybf);

  // Projection: out[8192x1024] = Ybf * Wprj + b_proj
  gemm_bf16_wmma<1><<<dim3(CC / 128, MM / 128), 256, 0, stream>>>(
      Ybf, Wprj, b_proj, nullptr, nullptr, nullptr, out, CC, CC);
}
